// MoELayer_67619965108245
// MI455X (gfx1250) — matrix operand order repaired and checked by reading, not compiled, not run on hardware
//
#include <hip/hip_runtime.h>
#include <math.h>

#define BB 64
#define NN 55
#define DD 512
#define EE 8
#define LL 4
#define HH (4*DD)     // 2048
#define MM (BB*NN)    // 3520

// B-tile LDS padding: each lane's 16-bf16 chunk stored at 24-element (48B)
// stride -> lane i starts at bank (12*i) mod 64: conflict-free b128 reads.
#define BLANE 24
#define BTILE (32 * BLANE)   // 768 elems per 32x16 tile in LDS

typedef __attribute__((ext_vector_type(16))) __bf16 v16bf;
typedef __attribute__((ext_vector_type(8)))  float  v8f;
typedef __attribute__((ext_vector_type(4)))  int    v4i;

typedef __attribute__((address_space(1))) v4i* gv4i_p;   // global
typedef __attribute__((address_space(3))) v4i* lv4i_p;   // LDS

union FragAB { v16bf v; uint4 q[2]; };

#if __has_builtin(__builtin_amdgcn_global_load_async_to_lds_b128)
#define USE_ASYNC 1
#else
#define USE_ASYNC 0
#endif

#if USE_ASYNC
__device__ __forceinline__ void cp16_async(void* lds, const void* g) {
  __builtin_amdgcn_global_load_async_to_lds_b128((gv4i_p)g, (lv4i_p)lds, 0, 0);
}
__device__ __forceinline__ void wait_async() {
#if __has_builtin(__builtin_amdgcn_s_wait_asynccnt)
  __builtin_amdgcn_s_wait_asynccnt(0);
#else
  asm volatile("s_wait_asynccnt 0x0" ::: "memory");
#endif
}
#endif

__device__ __forceinline__ __bf16 f2bf(float f) {
  unsigned u = __builtin_bit_cast(unsigned, f);
  unsigned r = (u + 0x7FFFu + ((u >> 16) & 1u)) >> 16;
  unsigned short s = (unsigned short)r;
  return __builtin_bit_cast(__bf16, s);
}

__device__ __forceinline__ float wave_sum(float v) {
  for (int o = 16; o > 0; o >>= 1) v += __shfl_xor(v, o, 32);
  return v;
}

// ---------------------------------------------------------------------------
// Pack fp32 weight matrices [C][K][Nn] (row-major, C = L*E matrices) into bf16
// WMMA B-fragment order: per 32x16 (KxN) tile, 32 lanes x 16 bf16 contiguous.
// Lane<16 : n = lane,     holds K = kt*32 + 0..15
// Lane>=16: n = lane-16,  holds K = kt*32 + 16..31
// ---------------------------------------------------------------------------
__global__ void pack_b_kernel(const float* __restrict__ src, __bf16* __restrict__ dst,
                              int K, int Nn, long total) {
  long id = (long)blockIdx.x * 256 + threadIdx.x;
  if (id >= total) return;
  long perMat = (long)K * Nn;
  int  c      = (int)(id / perMat);
  int  rem    = (int)(id % perMat);
  int  tile   = rem >> 9;       // /512 elems per tile
  int  within = rem & 511;
  int  ln     = within >> 4;    // lane 0..31
  int  j      = within & 15;    // element within lane
  int  ntN    = Nn >> 4;
  int  kt     = tile / ntN;
  int  nt     = tile % ntN;
  int  n      = (nt << 4) + (ln & 15);
  int  k      = (kt << 5) + ((ln >> 4) << 4) + j;
  dst[id] = f2bf(src[(size_t)c * perMat + (size_t)k * Nn + n]);
}

// ---------------------------------------------------------------------------
// Router: logits[b,n,e] = x[b,n,:] . Wr[n,e,:] + br[n,e]; softmax over E.
// ---------------------------------------------------------------------------
__global__ __launch_bounds__(256) void router_kernel(const float* __restrict__ x,
    const float* __restrict__ Wr, const float* __restrict__ br,
    float* __restrict__ wts) {
  int m = blockIdx.x;          // token index = b*N + n
  int n = m % NN;
  int e = threadIdx.x >> 5, ln = threadIdx.x & 31;
  __shared__ float lg[EE];
  const float* xr = x + (size_t)m * DD;
  const float* wr = Wr + ((size_t)n * EE + e) * DD;
  float s = 0.f;
  for (int j = ln; j < DD; j += 32) s += xr[j] * wr[j];
  s = wave_sum(s);
  if (ln == 0) lg[e] = s + br[n * EE + e];
  __syncthreads();
  if (threadIdx.x == 0) {
    float mx = lg[0];
    for (int i = 1; i < EE; ++i) mx = fmaxf(mx, lg[i]);
    float ex[EE]; float den = 0.f;
    for (int i = 0; i < EE; ++i) { ex[i] = __expf(lg[i] - mx); den += ex[i]; }
    float inv = 1.f / den;
    for (int i = 0; i < EE; ++i) wts[(size_t)m * EE + i] = ex[i] * inv;
  }
}

// h[e,m,d] = x[m,d]
__global__ void init_h_kernel(const float* __restrict__ x, float* __restrict__ h) {
  long total4 = (long)EE * MM * DD / 4;
  long i = (long)blockIdx.x * 256 + threadIdx.x;
  if (i >= total4) return;
  const float4* x4 = (const float4*)x;
  float4* h4 = (float4*)h;
  long per = (long)MM * DD / 4;
  h4[i] = x4[i % per];
}

// rmsnorm over D per (e,m) row -> bf16. One wave per row, 8 rows per block.
__global__ __launch_bounds__(256) void rmsnorm_kernel(const float* __restrict__ h,
    const float* __restrict__ scale, __bf16* __restrict__ nrm) {
  int row = blockIdx.x * 8 + (threadIdx.x >> 5);   // [0, E*M)
  int ln  = threadIdx.x & 31;
  int e   = row / MM;
  const float* hr = h + (size_t)row * DD;
  const float* sc = scale + (size_t)e * DD;
  float vals[16];
  float s = 0.f;
  for (int j = 0; j < 16; ++j) { float v = hr[ln + 32 * j]; vals[j] = v; s += v * v; }
  s = wave_sum(s);
  float inv = rsqrtf(s * (1.0f / DD) + 1e-8f);
  __bf16* o = nrm + (size_t)row * DD;
  for (int j = 0; j < 16; ++j) {
    int d = ln + 32 * j;
    o[d] = f2bf(vals[j] * inv * sc[d]);
  }
}

// ---------------------------------------------------------------------------
// GEMM1 + SwiGLU: gate = nrm@Wg, val = nrm@Wv (shared A tile),
// t = bf16(silu(gate)*val). Block tile 64x128, 8 waves of 2x2 WMMA tiles,
// double-buffered LDS, async global->LDS staging when available.
// ---------------------------------------------------------------------------
__global__ __launch_bounds__(256) void gemm1_swiglu_kernel(
    const __bf16* __restrict__ A,    // nrm [E, M, D]
    const __bf16* __restrict__ Bg,   // packed, layer-offset: [E][(D/32)*(H/16)][512]
    const __bf16* __restrict__ Bv,
    __bf16* __restrict__ T)          // [E, M, H]
{
  __shared__ __align__(16) __bf16 As[2][64 * 40];
  __shared__ __align__(16) __bf16 Bs[2][2][8 * BTILE];

  const int e    = blockIdx.z;
  const int m0   = blockIdx.x << 6;
  const int nblk = blockIdx.y;               // 0..15 (H/128)
  const int tid  = threadIdx.x;
  const int ln   = tid & 31;
  const int w    = tid >> 5;
  const int wm   = w >> 2, wn = w & 3;

  const __bf16* Ae  = A + ((size_t)e * MM + m0) * DD;
  const size_t tilesPerMat = (size_t)(DD / 32) * (HH / 16);
  const __bf16* BgE = Bg + (size_t)e * tilesPerMat * 512;
  const __bf16* BvE = Bv + (size_t)e * tilesPerMat * 512;

  auto copyA = [&](int s, int kt) {
    int row = tid >> 2, c4 = (tid & 3) << 3;
    const __bf16* src = Ae + (size_t)row * DD + (kt << 5) + c4;
    __bf16* dst = &As[s][row * 40 + c4];
#if USE_ASYNC
    cp16_async(dst, src);
#else
    *(uint4*)dst = *(const uint4*)src;
#endif
  };
  auto copyB = [&](int s, int kt) {
    size_t off = ((size_t)kt * (HH / 16) + (size_t)nblk * 8) * 512 + (size_t)tid * 16;
    int dsoff = (tid >> 5) * BTILE + (tid & 31) * BLANE;
#if USE_ASYNC
    cp16_async(&Bs[s][0][dsoff],     BgE + off);
    cp16_async(&Bs[s][0][dsoff + 8], BgE + off + 8);
    cp16_async(&Bs[s][1][dsoff],     BvE + off);
    cp16_async(&Bs[s][1][dsoff + 8], BvE + off + 8);
#else
    *(uint4*)&Bs[s][0][dsoff]     = *(const uint4*)(BgE + off);
    *(uint4*)&Bs[s][0][dsoff + 8] = *(const uint4*)(BgE + off + 8);
    *(uint4*)&Bs[s][1][dsoff]     = *(const uint4*)(BvE + off);
    *(uint4*)&Bs[s][1][dsoff + 8] = *(const uint4*)(BvE + off + 8);
#endif
  };

  v8f accG[2][2] = {}; v8f accV[2][2] = {};
  copyA(0, 0); copyB(0, 0);
#if USE_ASYNC
  wait_async();
#endif
  __syncthreads();

  const int KT = DD / 32;   // 16
  for (int kt = 0; kt < KT; ++kt) {
    int s = kt & 1;
    if (kt + 1 < KT) { copyA(s ^ 1, kt + 1); copyB(s ^ 1, kt + 1); }

    FragAB a[2], bg[2], bv[2];
    for (int i = 0; i < 2; ++i) {
      int row = wm * 32 + i * 16 + (ln & 15);
      int cb  = (ln >> 4) << 4;
      int off = row * 40 + cb;
      a[i].q[0] = *(const uint4*)&As[s][off];
      a[i].q[1] = *(const uint4*)&As[s][off + 8];
    }
    for (int j = 0; j < 2; ++j) {
      int off = (wn * 2 + j) * BTILE + ln * BLANE;
      bg[j].q[0] = *(const uint4*)&Bs[s][0][off];
      bg[j].q[1] = *(const uint4*)&Bs[s][0][off + 8];
      bv[j].q[0] = *(const uint4*)&Bs[s][1][off];
      bv[j].q[1] = *(const uint4*)&Bs[s][1][off + 8];
    }
    for (int i = 0; i < 2; ++i)
      for (int j = 0; j < 2; ++j) {
        accG[i][j] = __builtin_amdgcn_wmma_f32_16x16x32_bf16(
            false, a[i].v, false, bg[j].v, (short)0, accG[i][j], false, false);
        accV[i][j] = __builtin_amdgcn_wmma_f32_16x16x32_bf16(
            false, a[i].v, false, bv[j].v, (short)0, accV[i][j], false, false);
      }
#if USE_ASYNC
    if (kt + 1 < KT) wait_async();
#endif
    __syncthreads();
  }

  for (int i = 0; i < 2; ++i)
    for (int j = 0; j < 2; ++j) {
      int col = (nblk << 7) + wn * 32 + j * 16 + (ln & 15);
      int rb  = m0 + wm * 32 + i * 16 + ((ln >> 4) << 3);
      for (int r = 0; r < 8; ++r) {
        float g = accG[i][j][r], v = accV[i][j][r];
        float t = g / (1.f + __expf(-g)) * v;     // silu(g) * v
        T[((size_t)e * MM + rb + r) * HH + col] = f2bf(t);
      }
    }
}

// ---------------------------------------------------------------------------
// GEMM2: h += t @ Wo. Same tiling, single accumulator set, K = H = 2048.
// ---------------------------------------------------------------------------
__global__ __launch_bounds__(256) void gemm2_kernel(
    const __bf16* __restrict__ A,    // t [E, M, H]
    const __bf16* __restrict__ Bo,   // packed, layer-offset: [E][(H/32)*(D/16)][512]
    float* __restrict__ h)           // [E, M, D]
{
  __shared__ __align__(16) __bf16 As[2][64 * 40];
  __shared__ __align__(16) __bf16 Bs[2][8 * BTILE];

  const int e    = blockIdx.z;
  const int m0   = blockIdx.x << 6;
  const int nblk = blockIdx.y;               // 0..3 (D/128)
  const int tid  = threadIdx.x;
  const int ln   = tid & 31;
  const int w    = tid >> 5;
  const int wm   = w >> 2, wn = w & 3;

  const __bf16* Ae  = A + ((size_t)e * MM + m0) * HH;
  const size_t tilesPerMat = (size_t)(HH / 32) * (DD / 16);
  const __bf16* BoE = Bo + (size_t)e * tilesPerMat * 512;

  auto copyA = [&](int s, int kt) {
    int row = tid >> 2, c4 = (tid & 3) << 3;
    const __bf16* src = Ae + (size_t)row * HH + (kt << 5) + c4;
    __bf16* dst = &As[s][row * 40 + c4];
#if USE_ASYNC
    cp16_async(dst, src);
#else
    *(uint4*)dst = *(const uint4*)src;
#endif
  };
  auto copyB = [&](int s, int kt) {
    size_t off = ((size_t)kt * (DD / 16) + (size_t)nblk * 8) * 512 + (size_t)tid * 16;
    int dsoff = (tid >> 5) * BTILE + (tid & 31) * BLANE;
#if USE_ASYNC
    cp16_async(&Bs[s][dsoff],     BoE + off);
    cp16_async(&Bs[s][dsoff + 8], BoE + off + 8);
#else
    *(uint4*)&Bs[s][dsoff]     = *(const uint4*)(BoE + off);
    *(uint4*)&Bs[s][dsoff + 8] = *(const uint4*)(BoE + off + 8);
#endif
  };

  v8f acc[2][2] = {};
  copyA(0, 0); copyB(0, 0);
#if USE_ASYNC
  wait_async();
#endif
  __syncthreads();

  const int KT = HH / 32;   // 64
  for (int kt = 0; kt < KT; ++kt) {
    int s = kt & 1;
    if (kt + 1 < KT) { copyA(s ^ 1, kt + 1); copyB(s ^ 1, kt + 1); }

    FragAB a[2], b[2];
    for (int i = 0; i < 2; ++i) {
      int row = wm * 32 + i * 16 + (ln & 15);
      int cb  = (ln >> 4) << 4;
      int off = row * 40 + cb;
      a[i].q[0] = *(const uint4*)&As[s][off];
      a[i].q[1] = *(const uint4*)&As[s][off + 8];
    }
    for (int j = 0; j < 2; ++j) {
      int off = (wn * 2 + j) * BTILE + ln * BLANE;
      b[j].q[0] = *(const uint4*)&Bs[s][off];
      b[j].q[1] = *(const uint4*)&Bs[s][off + 8];
    }
    for (int i = 0; i < 2; ++i)
      for (int j = 0; j < 2; ++j)
        acc[i][j] = __builtin_amdgcn_wmma_f32_16x16x32_bf16(
            false, a[i].v, false, b[j].v, (short)0, acc[i][j], false, false);
#if USE_ASYNC
    if (kt + 1 < KT) wait_async();
#endif
    __syncthreads();
  }

  for (int i = 0; i < 2; ++i)
    for (int j = 0; j < 2; ++j) {
      int col = (nblk << 7) + wn * 32 + j * 16 + (ln & 15);
      int rb  = m0 + wm * 32 + i * 16 + ((ln >> 4) << 3);
      for (int r = 0; r < 8; ++r) {
        float* hp = h + ((size_t)e * MM + rb + r) * DD + col;
        *hp += acc[i][j][r];
      }
    }
}

// ---------------------------------------------------------------------------
// expert_out[e,m] = h[e,m,:].Wp[e] + bp[e]; out[m] = sum_e expert_out*wts.
// ---------------------------------------------------------------------------
__global__ __launch_bounds__(256) void final_kernel(const float* __restrict__ h,
    const float* __restrict__ Wp, const float* __restrict__ bp,
    const float* __restrict__ wts, float* __restrict__ out) {
  int m = blockIdx.x;
  int e = threadIdx.x >> 5, ln = threadIdx.x & 31;
  __shared__ float part[EE];
  const float* hr = h + ((size_t)e * MM + m) * DD;
  const float* wp = Wp + (size_t)e * DD;
  float s = 0.f;
  for (int j = ln; j < DD; j += 32) s += hr[j] * wp[j];
  s = wave_sum(s);
  if (ln == 0) part[e] = (s + bp[e]) * wts[(size_t)m * EE + e];
  __syncthreads();
  if (threadIdx.x == 0) {
    float acc = 0.f;
    for (int i = 0; i < EE; ++i) acc += part[i];
    out[m] = acc;
  }
}

// ---------------------------------------------------------------------------
extern "C" void kernel_launch(void* const* d_in, const int* in_sizes, int n_in,
                              void* d_out, int out_size, void* d_ws, size_t ws_size,
                              hipStream_t stream) {
  (void)in_sizes; (void)n_in; (void)out_size; (void)ws_size;

  const float* x     = (const float*)d_in[0];
  const float* scale = (const float*)d_in[1];
  const float* Wg    = (const float*)d_in[2];
  const float* Wv    = (const float*)d_in[3];
  const float* Wo    = (const float*)d_in[4];
  const float* Wp    = (const float*)d_in[5];
  const float* bp    = (const float*)d_in[6];
  const float* Wr    = (const float*)d_in[7];
  const float* br    = (const float*)d_in[8];
  float* out = (float*)d_out;

  char* ws = (char*)d_ws;
  size_t off = 0;
  auto alloc = [&](size_t bytes) -> void* {
    off = (off + 255) & ~(size_t)255;
    void* p = ws + off;
    off += bytes;
    return p;
  };
  __bf16* wgP = (__bf16*)alloc((size_t)LL * EE * DD * HH * 2);
  __bf16* wvP = (__bf16*)alloc((size_t)LL * EE * DD * HH * 2);
  __bf16* woP = (__bf16*)alloc((size_t)LL * EE * HH * DD * 2);
  float*  h   = (float*) alloc((size_t)EE * MM * DD * 4);
  __bf16* nrm = (__bf16*)alloc((size_t)EE * MM * DD * 2);
  __bf16* t   = (__bf16*)alloc((size_t)EE * MM * HH * 2);
  float*  wts = (float*) alloc((size_t)MM * EE * 4);

  long totalW = (long)LL * EE * DD * HH;
  int  pb     = (int)((totalW + 255) / 256);
  pack_b_kernel<<<pb, 256, 0, stream>>>(Wg, wgP, DD, HH, totalW);
  pack_b_kernel<<<pb, 256, 0, stream>>>(Wv, wvP, DD, HH, totalW);
  pack_b_kernel<<<pb, 256, 0, stream>>>(Wo, woP, HH, DD, totalW);

  router_kernel<<<MM, 256, 0, stream>>>(x, Wr, br, wts);

  long tot4 = (long)EE * MM * DD / 4;
  init_h_kernel<<<(int)((tot4 + 255) / 256), 256, 0, stream>>>(x, h);

  for (int l = 0; l < LL; ++l) {
    rmsnorm_kernel<<<EE * MM / 8, 256, 0, stream>>>(h, scale + (size_t)l * EE * DD, nrm);
    dim3 g1(MM / 64, HH / 128, EE);
    gemm1_swiglu_kernel<<<g1, 256, 0, stream>>>(
        nrm, wgP + (size_t)l * EE * DD * HH, wvP + (size_t)l * EE * DD * HH, t);
    dim3 g2(MM / 64, DD / 128, EE);
    gemm2_kernel<<<g2, 256, 0, stream>>>(t, woP + (size_t)l * EE * HH * DD, h);
  }

  final_kernel<<<MM, 256, 0, stream>>>(h, Wp, bp, wts, out);
}
